// Policy_GAE_63574105915523
// MI455X (gfx1250) — matrix operand order repaired and checked by reading, not compiled, not run on hardware
//
#include <hip/hip_runtime.h>
#include <math.h>

#define NN        10000
#define EE        320000
#define DIN       128
#define DHID      128
#define DOUT      64
#define ROW_TILES 625            // 10000 / 16
#define SIM_CG    157            // ceil(625 / 4) col-groups of 4 tiles (16x64 per wave)

typedef __attribute__((ext_vector_type(16))) __bf16 v16bf;
typedef __attribute__((ext_vector_type(8)))  __bf16 v8bf;
typedef __attribute__((ext_vector_type(8)))  float  v8f;

// ---------------- WMMA operand loaders (CDNA5 16x16x32 bf16 layouts) ----------------
// A (16x32, MxK): lane = hi*16 + r holds row r; element j -> k = ((j>>3)<<4) + hi*8 + (j&7)
__device__ __forceinline__ v16bf load_a_f32(const float* __restrict__ base, int ld, int kb, int lane) {
    int r = lane & 15, hi = lane >> 4;
    const float* p = base + (size_t)r * ld + kb + hi * 8;
    v16bf a;
#pragma unroll
    for (int j = 0; j < 8; ++j) a[j]     = (__bf16)p[j];
#pragma unroll
    for (int j = 0; j < 8; ++j) a[j + 8] = (__bf16)p[16 + j];
    return a;
}

__device__ __forceinline__ v16bf load_a_bf16(const __bf16* __restrict__ base, int ld, int kb, int lane) {
    int r = lane & 15, hi = lane >> 4;
    const __bf16* p = base + (size_t)r * ld + kb + hi * 8;
    v8bf lo = *(const v8bf*)p;          // k = kb+hi*8    .. +7   (16B aligned)
    v8bf hh = *(const v8bf*)(p + 16);   // k = kb+16+hi*8 .. +7
    return __builtin_shufflevector(lo, hh, 0,1,2,3,4,5,6,7,8,9,10,11,12,13,14,15);
}

// B (32x16, KxN), source stored transposed as [n][K] bf16 (k contiguous):
// lane = hi*16 + n holds column n; element j -> k = kb + ((j>>3)<<4) + hi*8 + (j&7)
__device__ __forceinline__ v16bf load_b_bf16(const __bf16* __restrict__ Wt, int ldk, int kb, int lane) {
    int n = lane & 15, hi = lane >> 4;
    const __bf16* p = Wt + (size_t)n * ldk + kb + hi * 8;
    v8bf lo = *(const v8bf*)p;
    v8bf hh = *(const v8bf*)(p + 16);
    return __builtin_shufflevector(lo, hh, 0,1,2,3,4,5,6,7,8,9,10,11,12,13,14,15);
}

__device__ __forceinline__ v8f wmma_bf16(v16bf a, v16bf b, v8f c) {
    return __builtin_amdgcn_wmma_f32_16x16x32_bf16(false, a, false, b, (short)0, c, false, false);
}

// fast sigmoid for the 1e8-element sim-matrix epilogue: v_exp_f32 + v_rcp_f32
__device__ __forceinline__ float fast_sigmoid(float x) {
    return __builtin_amdgcn_rcpf(1.0f + __expf(-x));
}

// ---------------- Kernels ----------------
__global__ void zero_kernel(float* agg0, float* agg1, float* acc) {
    int i = blockIdx.x * blockDim.x + threadIdx.x;     // 1,280,000 threads exactly
    agg0[i] = 0.0f;
    agg1[i] = 0.0f;
    if (i < 2) acc[i] = 0.0f;
}

// transpose + f32->bf16 convert the four weight matrices; Wt is [n][K=128]
__global__ void prep_weights(const float* Wr0, const float* Wo0, const float* Wr1, const float* Wo1,
                             __bf16* Tr0, __bf16* To0, __bf16* Tr1, __bf16* To1) {
    int tid = blockIdx.x * blockDim.x + threadIdx.x;
    if (tid < 16384) {
        int n = tid >> 7, k = tid & 127;
        Tr0[n * 128 + k] = (__bf16)Wr0[k * 128 + n];
    } else if (tid < 32768) {
        int t = tid - 16384; int n = t >> 7, k = t & 127;
        To0[n * 128 + k] = (__bf16)Wo0[k * 128 + n];
    } else if (tid < 40960) {
        int t = tid - 32768; int n = t >> 7, k = t & 127;
        Tr1[n * 128 + k] = (__bf16)Wr1[k * 64 + n];
    } else if (tid < 49152) {
        int t = tid - 40960; int n = t >> 7, k = t & 127;
        To1[n * 128 + k] = (__bf16)Wo1[k * 64 + n];
    }
}

// one wave per edge: agg[dst] += feat[src], 128 dims (lane handles a float4)
__global__ void scatter_add(const float* __restrict__ feat, const int* __restrict__ idx,
                            float* __restrict__ agg) {
    int gid  = blockIdx.x * blockDim.x + threadIdx.x;
    int e    = gid >> 5;
    int lane = gid & 31;
    if (e >= EE) return;
    int s = idx[e], d = idx[EE + e];
    const float4 v = *(const float4*)(feat + (size_t)s * 128 + lane * 4);
    float* p = agg + (size_t)d * 128 + lane * 4;
    atomicAdd(p + 0, v.x);
    atomicAdd(p + 1, v.y);
    atomicAdd(p + 2, v.z);
    atomicAdd(p + 3, v.w);
}

// out = act(Aagg @ WtRel^T + Ax @ WtRoot^T + bias); one wave per 16x16 tile
__global__ void layer_gemm(const float* __restrict__ Aagg, const float* __restrict__ Ax,
                           const __bf16* __restrict__ WtRel, const __bf16* __restrict__ WtRoot,
                           const float* __restrict__ bias, float* __restrict__ outp,
                           int Ntot, int K, int doTanh) {
    int lane   = threadIdx.x & 31;
    int wid    = (blockIdx.x * blockDim.x + threadIdx.x) >> 5;
    int ntiles = Ntot >> 4;
    if (wid >= ROW_TILES * ntiles) return;          // wave-uniform
    int rt = wid / ntiles, nt = wid - rt * ntiles;

    const float*  arow = Aagg   + (size_t)rt * 16 * K;
    const float*  xrow = Ax     + (size_t)rt * 16 * K;
    const __bf16* wr   = WtRel  + (size_t)nt * 16 * K;
    const __bf16* wo   = WtRoot + (size_t)nt * 16 * K;

    v8f c = {};
    for (int kb = 0; kb < K; kb += 32) {
        v16bf a0 = load_a_f32(arow, K, kb, lane);
        v16bf b0 = load_b_bf16(wr, K, kb, lane);
        c = wmma_bf16(a0, b0, c);
        v16bf a1 = load_a_f32(xrow, K, kb, lane);
        v16bf b1 = load_b_bf16(wo, K, kb, lane);
        c = wmma_bf16(a1, b1, c);
    }

    int n = lane & 15, hi = lane >> 4;
    int col = nt * 16 + n;
    float bv = bias[col];
#pragma unroll
    for (int v = 0; v < 8; ++v) {
        int row = rt * 16 + v + 8 * hi;
        float val = c[v] + bv;
        outp[(size_t)row * Ntot + col] = doTanh ? tanhf(val) : val;
    }
}

// row-normalize h2 and convert to bf16 (one wave per node)
__global__ void norm_kernel(const float* __restrict__ h2, __bf16* __restrict__ hn) {
    int wid  = (blockIdx.x * blockDim.x + threadIdx.x) >> 5;
    int lane = threadIdx.x & 31;
    if (wid >= NN) return;
    const float* p = h2 + (size_t)wid * 64;
    float a = p[lane], b = p[lane + 32];
    float ss = a * a + b * b;
#pragma unroll
    for (int m = 16; m >= 1; m >>= 1) ss += __shfl_xor(ss, m, 32);
    float inv = 1.0f / fmaxf(sqrtf(ss), 1e-8f);
    hn[(size_t)wid * 64 + lane]      = (__bf16)(a * inv);
    hn[(size_t)wid * 64 + lane + 32] = (__bf16)(b * inv);
}

// sim = sigmoid(hn @ hn^T): one wave computes a 16x64 strip (A reused across 4 B tiles)
__global__ void sim_kernel(const __bf16* __restrict__ hn, float* __restrict__ outp) {
    int lane = threadIdx.x & 31;
    int wid  = (blockIdx.x * blockDim.x + threadIdx.x) >> 5;
    if (wid >= ROW_TILES * SIM_CG) return;          // wave-uniform
    int rt = wid / SIM_CG, cg = wid - rt * SIM_CG;
    int ct0 = cg * 4;

    const __bf16* arow = hn + (size_t)rt * 16 * DOUT;
    v8f c[4] = {};
#pragma unroll
    for (int kb = 0; kb < DOUT; kb += 32) {
        v16bf a = load_a_bf16(arow, DOUT, kb, lane);
#pragma unroll
        for (int t = 0; t < 4; ++t) {
            if (ct0 + t < ROW_TILES) {              // wave-uniform
                v16bf b = load_b_bf16(hn + (size_t)(ct0 + t) * 16 * DOUT, DOUT, kb, lane);
                c[t] = wmma_bf16(a, b, c[t]);
            }
        }
    }

    int n = lane & 15, hi = lane >> 4;
#pragma unroll
    for (int t = 0; t < 4; ++t) {
        int ct = ct0 + t;
        if (ct < ROW_TILES) {
            int col = ct * 16 + n;
#pragma unroll
            for (int v = 0; v < 8; ++v) {
                int row = rt * 16 + v + 8 * hi;
                outp[(size_t)row * NN + col] = fast_sigmoid(c[t][v]);
            }
        }
    }
}

// BCE edge loss: one wave per edge batch-of-32; block LDS reduce -> 1 atomic/block
__global__ void edge_loss_kernel(const float* __restrict__ h2, const int* __restrict__ idx,
                                 float* __restrict__ acc, int negFlag) {
    __shared__ float sred[8];
    int lane = threadIdx.x & 31;
    int wib  = threadIdx.x >> 5;
    int wid  = (blockIdx.x * blockDim.x + threadIdx.x) >> 5;
    int nw   = (gridDim.x * blockDim.x) >> 5;
    float local = 0.0f;
    for (int e = wid; e < EE; e += nw) {
        int s = idx[e], d = idx[EE + e];
        const float* ps = h2 + (size_t)s * 64;
        const float* pd = h2 + (size_t)d * 64;
        float dot = ps[lane] * pd[lane] + ps[lane + 32] * pd[lane + 32];
#pragma unroll
        for (int m = 16; m >= 1; m >>= 1) dot += __shfl_xor(dot, m, 32);
        float z = negFlag ? dot : -dot;
        local += fmaxf(z, 0.0f) + log1pf(expf(-fabsf(z)));   // stable softplus(z)
    }
    if (lane == 0) sred[wib] = local;
    __syncthreads();
    if (threadIdx.x == 0) {
        float s = 0.0f;
#pragma unroll
        for (int i = 0; i < 8; ++i) s += sred[i];
        atomicAdd(&acc[negFlag], s);
    }
}

__global__ void finalize_kernel(const float* __restrict__ acc, float* __restrict__ outp) {
    if (blockIdx.x == 0 && threadIdx.x == 0) {
        outp[(size_t)NN * NN] = acc[0] * (1.0f / EE) + acc[1] * (1.0f / EE);
    }
}

// ---------------- Host launcher ----------------
extern "C" void kernel_launch(void* const* d_in, const int* in_sizes, int n_in,
                              void* d_out, int out_size, void* d_ws, size_t ws_size,
                              hipStream_t stream) {
    (void)in_sizes; (void)n_in; (void)out_size; (void)ws_size;
    const float* x   = (const float*)d_in[1];
    const int*   ei  = (const int*)d_in[2];
    const int*   nei = (const int*)d_in[3];
    const float* Wr0 = (const float*)d_in[4];
    const float* Wo0 = (const float*)d_in[5];
    const float* b0  = (const float*)d_in[6];
    const float* Wr1 = (const float*)d_in[7];
    const float* Wo1 = (const float*)d_in[8];
    const float* b1  = (const float*)d_in[9];
    float* out = (float*)d_out;

    char* ws = (char*)d_ws;                         // all offsets 256B-aligned
    float*  agg0 = (float*)(ws + 0);                //  NN*128 f32 = 5,120,000 B
    float*  h    = (float*)(ws + 5120000);          //  NN*128 f32
    float*  agg1 = (float*)(ws + 10240000);         //  NN*128 f32
    float*  h2   = (float*)(ws + 15360000);         //  NN*64  f32 = 2,560,000 B
    __bf16* hn   = (__bf16*)(ws + 17920000);        //  NN*64  bf16 = 1,280,000 B
    __bf16* Tr0  = (__bf16*)(ws + 19200000);        //  128x128 bf16
    __bf16* To0  = (__bf16*)(ws + 19232768);
    __bf16* Tr1  = (__bf16*)(ws + 19265536);        //  64x128 bf16
    __bf16* To1  = (__bf16*)(ws + 19281920);
    float*  acc  = (float*)(ws + 19298304);         //  2 f32

    zero_kernel<<<5000, 256, 0, stream>>>(agg0, agg1, acc);                       // 1.28M threads
    prep_weights<<<192, 256, 0, stream>>>(Wr0, Wo0, Wr1, Wo1, Tr0, To0, Tr1, To1);

    scatter_add<<<(EE * 32) / 256, 256, 0, stream>>>(x, ei, agg0);
    layer_gemm<<<625, 256, 0, stream>>>(agg0, x, Tr0, To0, b0, h, DHID, DIN, 1);  // 5000 waves

    scatter_add<<<(EE * 32) / 256, 256, 0, stream>>>(h, ei, agg1);
    layer_gemm<<<313, 256, 0, stream>>>(agg1, h, Tr1, To1, b1, h2, DOUT, DHID, 0);// 2500 waves

    norm_kernel<<<1250, 256, 0, stream>>>(h2, hn);                                // 10000 waves
    sim_kernel<<<12266, 256, 0, stream>>>(hn, out);                               // 98125 waves

    edge_loss_kernel<<<1250, 256, 0, stream>>>(h2, ei, acc, 0);
    edge_loss_kernel<<<1250, 256, 0, stream>>>(h2, nei, acc, 1);
    finalize_kernel<<<1, 64, 0, stream>>>(acc, out);
}
